// BinReLUConvBN_43868795962037
// MI455X (gfx1250) — matrix operand order repaired and checked by reading, not compiled, not run on hardware
//
#include <hip/hip_runtime.h>
#include <hip/hip_bf16.h>

typedef __attribute__((ext_vector_type(16))) _Float16 v16h;
typedef __attribute__((ext_vector_type(8)))  _Float16 v8h;
typedef __attribute__((ext_vector_type(4)))  _Float16 v4h;
typedef __attribute__((ext_vector_type(2)))  _Float16 v2h;
typedef __attribute__((ext_vector_type(8)))  float    v8f;

#define Nn 64
#define Cc 512
#define Hh 32
#define Ww 32
#define HW 1024           // 32*32
#define GROUPS 16
#define CIG 32            // channels in per group
#define KTAPS 9           // 3x3
#define KPG 288           // CIG * 9
#define NHW 65536         // Nn*HW (per-channel element count)

// ---------------------------------------------------------------------------
// Per-channel sum / sum-of-squares over NCHW. T = float (x / f32 conv) or
// _Float16 (raw integer conv sums stored exactly in f16).
// ---------------------------------------------------------------------------
template <typename T>
__global__ __launch_bounds__(256) void chan_stats(const T* __restrict__ src,
                                                  float* __restrict__ sum,
                                                  float* __restrict__ sumsq) {
  int c = blockIdx.x;
  int t = threadIdx.x;             // 256 threads, 4 elements each per n
  float s = 0.f, s2 = 0.f;
  for (int n = 0; n < Nn; ++n) {
    size_t base = ((size_t)(n * Cc + c)) * HW;
    if constexpr (sizeof(T) == 4) {
      float4 v = *((const float4*)(src + base) + t);
      s  += v.x + v.y + v.z + v.w;
      s2 += v.x * v.x + v.y * v.y + v.z * v.z + v.w * v.w;
    } else {
      v4h v = *((const v4h*)(src + base) + t);
      float a0 = (float)v[0], a1 = (float)v[1], a2 = (float)v[2], a3 = (float)v[3];
      s  += a0 + a1 + a2 + a3;
      s2 += a0 * a0 + a1 * a1 + a2 * a2 + a3 * a3;
    }
  }
  __shared__ float ls[256], ls2[256];
  ls[t] = s; ls2[t] = s2;
  __syncthreads();
  for (int o = 128; o > 0; o >>= 1) {
    if (t < o) { ls[t] += ls[t + o]; ls2[t] += ls2[t + o]; }
    __syncthreads();
  }
  if (t == 0) { sum[c] = ls[0]; sumsq[c] = ls2[0]; }
}

__global__ __launch_bounds__(512) void finalize_stats(const float* __restrict__ sum,
                                                      const float* __restrict__ sumsq,
                                                      float* __restrict__ mean,
                                                      float* __restrict__ rstd) {
  int c = threadIdx.x;             // 512 threads, one block
  float m = sum[c] * (1.f / (float)NHW);
  float v = sumsq[c] * (1.f / (float)NHW) - m * m;
  mean[c] = m;
  rstd[c] = rsqrtf(v + 1e-5f);
}

// BN2 fold: stats were taken on stored conv values; if USE_SW the stored
// values are raw integer sums and the per-channel weight scale sw folds in:
//   conv = sw*raw ; mean = sw*m_raw ; var = sw^2*var_raw
//   y = (conv-mean)*rsqrt(var+eps)*g + b  ==  raw*A + B
template <bool USE_SW>
__global__ __launch_bounds__(512) void finalize_ab(const float* __restrict__ sum,
                                                   const float* __restrict__ sumsq,
                                                   const float* __restrict__ sw,
                                                   const float* __restrict__ g2,
                                                   const float* __restrict__ b2,
                                                   float* __restrict__ A,
                                                   float* __restrict__ B) {
  int c = threadIdx.x;
  float m_raw = sum[c] * (1.f / (float)NHW);
  float v_raw = sumsq[c] * (1.f / (float)NHW) - m_raw * m_raw;
  float s = USE_SW ? sw[c] : 1.f;
  float m = s * m_raw;
  float var = s * s * v_raw;
  float rstd = rsqrtf(var + 1e-5f);
  A[c] = s * rstd * g2[c];
  B[c] = b2[c] - m * rstd * g2[c];
}

// ---------------------------------------------------------------------------
// BN1 + sign(), NCHW f32 -> NHWC f16 (LDS transpose, coalesced in and out)
// tile: 64 channels x 64 pixels per block, grid = 64n * 16pix * 8chan = 8192
// ---------------------------------------------------------------------------
__global__ __launch_bounds__(256) void binarize_transpose(const float* __restrict__ x,
                                                          const float* __restrict__ mean,
                                                          const float* __restrict__ rstd,
                                                          const float* __restrict__ gamma,
                                                          const float* __restrict__ beta,
                                                          _Float16* __restrict__ act) {
  int bid = blockIdx.x;
  int cb = bid & 7;
  int pb = (bid >> 3) & 15;
  int n  = bid >> 7;
  int c0 = cb * 64;
  int p0 = pb * 64;

  __shared__ _Float16 tile[64][65];

  for (int k = 0; k < 16; ++k) {
    int idx = threadIdx.x + k * 256;   // 0..4095
    int row = idx >> 6;                // channel in tile
    int col = idx & 63;                // pixel in tile
    int c = c0 + row;
    float v  = x[((size_t)(n * Cc + c)) * HW + p0 + col];
    float xh = (v - mean[c]) * rstd[c] * gamma[c] + beta[c];
    float sg = (xh > 0.f) ? 1.f : ((xh < 0.f) ? -1.f : 0.f);
    tile[row][col] = (_Float16)sg;
  }
  __syncthreads();
  for (int k = 0; k < 8; ++k) {
    int idx = threadIdx.x + k * 256;   // 0..2047
    int px  = idx >> 5;
    int ch2 = (idx & 31) * 2;
    v2h val = { tile[ch2][px], tile[ch2 + 1][px] };
    *(v2h*)(act + ((size_t)(n * HW + p0 + px)) * Cc + c0 + ch2) = val;
  }
}

// ---------------------------------------------------------------------------
// Weight prep: per output channel, compute sw = mean(|(w-mu)/std|) (ddof=1)
// and pack sign(w) into the WMMA B-operand lane layout:
//   wpack[(g*2+half)*9 + j][lane][e], lane<16 -> col n=lane, K(ci)=e
//                                     lane>=16 -> col n=lane-16, K(ci)=16+e
// ---------------------------------------------------------------------------
__global__ __launch_bounds__(288) void weight_prep(const float* __restrict__ w,
                                                   _Float16* __restrict__ wpack,
                                                   float* __restrict__ sw) {
  int co = blockIdx.x;
  int t  = threadIdx.x;              // 0..287 : t = ci*9 + kh*3 + kw
  float wv = w[(size_t)co * KPG + t];

  __shared__ float red[KPG];
  __shared__ float smean, sstd;
  red[t] = wv; __syncthreads();
  if (t == 0) { float s = 0.f; for (int i = 0; i < KPG; ++i) s += red[i]; smean = s / (float)KPG; }
  __syncthreads();
  float d = wv - smean;
  red[t] = d * d; __syncthreads();
  if (t == 0) { float s = 0.f; for (int i = 0; i < KPG; ++i) s += red[i]; sstd = sqrtf(s / (float)(KPG - 1)); }
  __syncthreads();
  red[t] = fabsf(d); __syncthreads();
  if (t == 0) { float s = 0.f; for (int i = 0; i < KPG; ++i) s += red[i]; sw[co] = s / ((float)KPG * sstd); }

  int ci = t / KTAPS;
  int j  = t % KTAPS;                // kh*3+kw
  int g = co >> 5, half = (co >> 4) & 1, ncol = co & 15;
  int lane = (ci < 16) ? ncol : (ncol + 16);
  int e = ci & 15;
  float sg = (wv > 0.f) ? 1.f : ((wv < 0.f) ? -1.f : 0.f);
  wpack[((size_t)(((g * 2 + half) * KTAPS + j) * 32 + lane)) * 16 + e] = (_Float16)sg;
}

// ---------------------------------------------------------------------------
// Grouped 3x3 conv as WMMA GEMM: per wave one 16(pixels) x 16(out-chan) tile,
// K=288 = 9 taps x 32 ci, one v_wmma_f32_16x16x32_f16 per tap.
// A layout (contiguous K, as wpack): lane<16 -> row M=lane, K 0-15
//                                lane>=16 -> row M=lane-16, K 16-31
// NHWC act makes both 8-half groups contiguous 16B loads.
// OUT16: store raw integer accumulator (exact in f16, |sum|<=288) to halve
// the downstream traffic; sw is folded into BN2 instead.
// ---------------------------------------------------------------------------
template <bool OUT16>
__global__ __launch_bounds__(256) void conv_wmma(const _Float16* __restrict__ act,
                                                 const _Float16* __restrict__ wpack,
                                                 const float* __restrict__ sw,
                                                 void* __restrict__ outv) {
  int wave = (int)((blockIdx.x * 256u + threadIdx.x) >> 5);
  int lane = threadIdx.x & 31;

  int coTile = wave & 31;            // 32 output-channel tiles (16 groups x 2)
  int mTile  = wave >> 5;            // 4096 pixel tiles
  int g = coTile >> 1, half = coTile & 1;

  int m0 = mTile << 4;
  int n  = m0 >> 10;
  int hw = m0 & 1023;
  int h  = hw >> 5;
  int w0 = hw & 31;                  // 0 or 16 (tiles never straddle a row)

  int lm    = lane & 15;             // A-row (pixel) index within tile
  int hiSel = lane >> 4;             // K-half selector

  const _Float16* wp = wpack + ((size_t)((g * 2 + half) * KTAPS)) * 512 + (size_t)lane * 16;

  v8f acc = {};
  for (int j = 0; j < KTAPS; ++j) {
    int kh = j / 3, kw = j % 3;
    int hh = h + kh - 1;
    int ww = w0 + lm + kw - 1;
    v16h a = {};
    if ((unsigned)hh < (unsigned)Hh && (unsigned)ww < (unsigned)Ww) {
      const _Float16* ap = act + ((size_t)(n * HW + hh * Ww + ww)) * Cc + g * CIG + hiSel * 16;
      v8h lo = *(const v8h*)ap;          // ci 0-7  (or 16-23)
      v8h hi = *(const v8h*)(ap + 8);    // ci 8-15 (or 24-31)
      a = __builtin_shufflevector(lo, hi, 0,1,2,3,4,5,6,7,8,9,10,11,12,13,14,15);
    }
    v16h b = *(const v16h*)(wp + (size_t)j * 512);
    acc = __builtin_amdgcn_wmma_f32_16x16x32_f16(false, a, false, b, (short)0, acc, false, false);
  }

  // D layout: VGPR r, lane<16 -> pixel m0+r, chan lane; lane>=16 -> pixel m0+8+r, chan lane-16
  int co = g * CIG + half * 16 + lm;
  size_t oidx = ((size_t)(n * Cc + co)) * HW + h * Ww + w0 + hiSel * 8;
  if constexpr (OUT16) {
    v8h h8;
    for (int k = 0; k < 8; ++k) h8[k] = (_Float16)acc[k];   // exact integers
    *(v8h*)((_Float16*)outv + oidx) = h8;
  } else {
    float scale = sw[co];
    float* op = (float*)outv + oidx;
    float4 s0 = make_float4(acc[0] * scale, acc[1] * scale, acc[2] * scale, acc[3] * scale);
    float4 s1 = make_float4(acc[4] * scale, acc[5] * scale, acc[6] * scale, acc[7] * scale);
    *(float4*)(op)     = s0;
    *(float4*)(op + 4) = s1;
  }
}

// ---------------------------------------------------------------------------
// Fused BN2 + residual add + PReLU:  h = conv*A[c] + B[c] + x ; PReLU(h)
// ---------------------------------------------------------------------------
template <typename T>
__global__ __launch_bounds__(256) void fuse_final(const float* __restrict__ x,
                                                  const T* __restrict__ conv,
                                                  float* __restrict__ out,
                                                  const float* __restrict__ A,
                                                  const float* __restrict__ B,
                                                  const float* __restrict__ pa) {
  size_t i = ((size_t)blockIdx.x * 256 + threadIdx.x) * 4;
  int c = (int)((i >> 10) & (Cc - 1));
  float a = A[c], b = B[c], al = pa[c];
  float c0, c1, c2, c3;
  if constexpr (sizeof(T) == 4) {
    float4 cv = *(const float4*)(conv + i);
    c0 = cv.x; c1 = cv.y; c2 = cv.z; c3 = cv.w;
  } else {
    v4h cv = *(const v4h*)(conv + i);
    c0 = (float)cv[0]; c1 = (float)cv[1]; c2 = (float)cv[2]; c3 = (float)cv[3];
  }
  const float4 xv = *(const float4*)(x + i);
  float4 r;
  float h;
  h = c0 * a + b + xv.x; r.x = h > 0.f ? h : al * h;
  h = c1 * a + b + xv.y; r.y = h > 0.f ? h : al * h;
  h = c2 * a + b + xv.z; r.z = h > 0.f ? h : al * h;
  h = c3 * a + b + xv.w; r.w = h > 0.f ? h : al * h;
  *(float4*)(out + i) = r;
}

// ---------------------------------------------------------------------------
extern "C" void kernel_launch(void* const* d_in, const int* in_sizes, int n_in,
                              void* d_out, int out_size, void* d_ws, size_t ws_size,
                              hipStream_t stream) {
  const float* x      = (const float*)d_in[0];
  const float* conv_w = (const float*)d_in[1];
  const float* g1     = (const float*)d_in[2];
  const float* b1     = (const float*)d_in[3];
  const float* g2     = (const float*)d_in[4];
  const float* b2     = (const float*)d_in[5];
  const float* pa     = (const float*)d_in[6];
  float* out = (float*)d_out;

  const size_t actB   = (size_t)Nn * HW * Cc * 2;              // 64 MB NHWC f16
  const size_t wpackB = (size_t)GROUPS * 2 * KTAPS * 512 * 2;  // 288 KB
  const size_t convB  = (size_t)Nn * Cc * HW * 2;              // 64 MB f16 conv

  char* ws = (char*)d_ws;
  size_t off = 0;
  _Float16* act   = (_Float16*)(ws + off); off += actB;
  _Float16* wpack = (_Float16*)(ws + off); off += wpackB;
  float* sw    = (float*)(ws + off); off += Cc * 4;
  float* sum1  = (float*)(ws + off); off += Cc * 4;
  float* sq1   = (float*)(ws + off); off += Cc * 4;
  float* mean1 = (float*)(ws + off); off += Cc * 4;
  float* rstd1 = (float*)(ws + off); off += Cc * 4;
  float* sum2  = (float*)(ws + off); off += Cc * 4;
  float* sq2   = (float*)(ws + off); off += Cc * 4;
  float* A2    = (float*)(ws + off); off += Cc * 4;
  float* B2    = (float*)(ws + off); off += Cc * 4;
  _Float16* conv16 = (_Float16*)(ws + off);
  const bool f16path = (off + convB) <= ws_size;   // deterministic: ws_size fixed

  // 1. BN1 stats on x
  chan_stats<float><<<Cc, 256, 0, stream>>>(x, sum1, sq1);
  finalize_stats<<<1, Cc, 0, stream>>>(sum1, sq1, mean1, rstd1);
  // 2. binarize + transpose to NHWC f16
  binarize_transpose<<<Nn * 16 * 8, 256, 0, stream>>>(x, mean1, rstd1, g1, b1, act);
  // 3. weight scale + sign-pack into WMMA B layout
  weight_prep<<<Cc, KPG, 0, stream>>>(conv_w, wpack, sw);

  const int convBlocks = (4096 * 32) / 8;
  const int fuseBlocks = (Nn * Cc * HW) / (4 * 256);
  if (f16path) {
    // 4. WMMA conv -> raw integer sums stored exactly as f16 (halves traffic)
    conv_wmma<true><<<convBlocks, 256, 0, stream>>>(act, wpack, sw, (void*)conv16);
    // 5. BN2 stats on raw sums; fold sw into per-channel affine (A,B)
    chan_stats<_Float16><<<Cc, 256, 0, stream>>>(conv16, sum2, sq2);
    finalize_ab<true><<<1, Cc, 0, stream>>>(sum2, sq2, sw, g2, b2, A2, B2);
    // 6. fused BN2 + residual + PReLU
    fuse_final<_Float16><<<fuseBlocks, 256, 0, stream>>>(x, conv16, out, A2, B2, pa);
  } else {
    // Fallback: f32 conv result lands in d_out, processed in place
    conv_wmma<false><<<convBlocks, 256, 0, stream>>>(act, wpack, sw, (void*)out);
    chan_stats<float><<<Cc, 256, 0, stream>>>(out, sum2, sq2);
    finalize_ab<false><<<1, Cc, 0, stream>>>(sum2, sq2, sw, g2, b2, A2, B2);
    fuse_final<float><<<fuseBlocks, 256, 0, stream>>>(x, out, out, A2, B2, pa);
  }
}
